// MDLSTM_48730698940887
// MI455X (gfx1250) — matrix operand order repaired and checked by reading, not compiled, not run on hardware
//
#include <hip/hip_runtime.h>
#include <hip/hip_bf16.h>

// MDLSTM fused dual-GEMM for MI455X (gfx1250, wave32, WMMA).
// out[m,n] = sigmoid(x@wo + bo)[m,n] * tanh(sigmoid(x@wc + bc)[m,n])
// M = 64*64*32 = 131072, K = 128, N = 256.
// f16 WMMA (f32 accumulate) -> memory-bound at ~200MB / 23.3 TB/s.
// Each wave owns 32 rows (2 M-tiles) so every B fragment read from LDS feeds
// two WMMAs: halves LDS B-traffic and gives 4 independent accumulator chains
// (O/C x m0/m1) that cover the f16 WMMA->WMMA RAW hazard (1 NOP + 4 coexec).
// Epilogue: CDNA5 v_tanh_f32 only (sigmoid via tanh identity), 3 TRANS/elem.

typedef __attribute__((ext_vector_type(16))) _Float16 v16h;
typedef __attribute__((ext_vector_type(8)))  float    v8f;

#define NCHUNKS 4096         // 131072 rows / 32 rows per wave
#define GRIDX   512          // 512 blocks x 8 waves = 4096 chunks, exact

#if __has_builtin(__builtin_amdgcn_tanhf)
__device__ __forceinline__ float hw_tanh(float x) { return __builtin_amdgcn_tanhf(x); }
#elif __has_builtin(__builtin_amdgcn_tanh_f32)
__device__ __forceinline__ float hw_tanh(float x) { return __builtin_amdgcn_tanh_f32(x); }
#else
__device__ __forceinline__ float hw_tanh(float x) {
  float e = __builtin_amdgcn_exp2f(x * -2.88539008177792681472f); // e^-2x
  return (1.0f - e) * __builtin_amdgcn_rcpf(1.0f + e);
}
#endif

#define WMMA_F16(A, B, C) \
  __builtin_amdgcn_wmma_f32_16x16x32_f16(false, (A), false, (B), (short)0, (C), false, false)

// ---------------------------------------------------------------------------
// Prep: pack wo (g=0) and wc (g=1), f32 (K=128 x N=256 row-major) -> f16 in
// the CDNA5 WMMA B-fragment layout for V_WMMA_F32_16X16X32_F16:
//   B is 32x16 (KxN); lane = N (lanes 0-15 => K 0..7,16..23, lanes 16-31 => K 8..15,24..31),
//   16 halves per lane, half j => K = kstep*32 + (lane<16 ? 0 : 8) + (j&7) + (j&8 ? 16 : 0).
// Packed index (halves): (((g*16 + ntile)*4 + kstep)*32 + lane)*16 + j
// Total: 2*16*4*32*16 = 65536 halves = 128 KB in d_ws.
// ---------------------------------------------------------------------------
__global__ void pack_weights_f16(const float* __restrict__ wo,
                                 const float* __restrict__ wc,
                                 _Float16* __restrict__ pk) {
  int idx = blockIdx.x * 256 + threadIdx.x;      // 0..65535
  int j = idx & 15;
  int l = (idx >> 4) & 31;
  int s = (idx >> 9) & 3;
  int t = (idx >> 11) & 15;
  int g = (idx >> 15) & 1;
  int n = t * 16 + (l & 15);
  int k = s * 32 + ((l < 16) ? 0 : 8) + (j & 7) + ((j & 8) ? 16 : 0);
  const float* w = g ? wc : wo;
  pk[idx] = (_Float16)w[k * 256 + n];
}

// A fragment: 16x32 f16 per k-step, per ISA layout:
// lanes 0-15 : halves 0-7 = K 0..7,  halves 8-15 = K 16..23
// lanes 16-31: halves 0-7 = K 8..15, halves 8-15 = K 24..31
__device__ __forceinline__ v16h load_a_frag(const float* __restrict__ xr,
                                            int s, int lane) {
  const int k0 = s * 32 + ((lane < 16) ? 0 : 8);
  float4 p0 = *(const float4*)(xr + k0);
  float4 p1 = *(const float4*)(xr + k0 + 4);
  float4 p2 = *(const float4*)(xr + k0 + 16);
  float4 p3 = *(const float4*)(xr + k0 + 20);
  v16h a;
  a[0]  = (_Float16)p0.x; a[1]  = (_Float16)p0.y;
  a[2]  = (_Float16)p0.z; a[3]  = (_Float16)p0.w;
  a[4]  = (_Float16)p1.x; a[5]  = (_Float16)p1.y;
  a[6]  = (_Float16)p1.z; a[7]  = (_Float16)p1.w;
  a[8]  = (_Float16)p2.x; a[9]  = (_Float16)p2.y;
  a[10] = (_Float16)p2.z; a[11] = (_Float16)p2.w;
  a[12] = (_Float16)p3.x; a[13] = (_Float16)p3.y;
  a[14] = (_Float16)p3.z; a[15] = (_Float16)p3.w;
  return a;
}

// ---------------------------------------------------------------------------
// Main fused kernel. 256 threads = 8 wave32 per block, 32 rows per wave.
// Weights (128 KB) + half-biases (2 KB) staged in LDS once per block
// (CDNA5 WGP: 320 KB -> 2 blocks/WGP resident).
// ---------------------------------------------------------------------------
__global__ void __launch_bounds__(256)
mdlstm_fused_wmma(const float* __restrict__ x,
                  const _Float16* __restrict__ pk,
                  const float* __restrict__ biaso,
                  const float* __restrict__ biasc,
                  float* __restrict__ out) {
  __shared__ _Float16 ldsW[65536];               // 128 KB packed B fragments
  __shared__ float    ldsB[512];                 // 0.5*biaso | 0.5*biasc

  // Cooperative LDS fill (b128 loads + ds_store_b128; biases as 0.5*b).
  {
    const uint4* __restrict__ src = (const uint4*)pk;
    uint4* dst = (uint4*)ldsW;
    for (int i = threadIdx.x; i < 8192; i += 256) dst[i] = src[i];
    int i = threadIdx.x;
    ldsB[i]       = 0.5f * biaso[i];
    ldsB[i + 256] = 0.5f * biasc[i];
  }
  __syncthreads();

  const int lane = threadIdx.x & 31;
  const int wv   = threadIdx.x >> 5;
  const int lh   = lane & 15;                    // row (A) / column (C,D)
  const int hi8  = (lane >= 16) ? 8 : 0;         // C/D: upper half-lanes = rows +8

  for (int ch = blockIdx.x * 8 + wv; ch < NCHUNKS; ch += gridDim.x * 8) {
    // --- Two A-fragment sets: rows [ch*32, ch*32+16) and [ch*32+16, ch*32+32)
    const float* __restrict__ xr0 = x + (size_t)(ch * 32 + lh) * 128;
    const float* __restrict__ xr1 = xr0 + 16 * 128;
    v16h a0[4], a1[4];
#pragma unroll
    for (int s = 0; s < 4; ++s) {
      a0[s] = load_a_frag(xr0, s, lane);
      a1[s] = load_a_frag(xr1, s, lane);
    }

    // --- Sweep 16 N-tiles; per tile: 4 k-steps x 2 gemms x 2 M-tiles =
    // 16 WMMA fed by only 8 B-fragment loads (each B reused for m0 and m1).
    for (int t = 0; t < 16; ++t) {
      const _Float16* __restrict__ wtO = ldsW + (size_t)t * 2048 + (size_t)lane * 16;
      const _Float16* __restrict__ wtC = wtO + 32768;   // g=1 block

      v8f acO0 = {}, acO1 = {}, acC0 = {}, acC1 = {};
#pragma unroll
      for (int s = 0; s < 4; ++s) {
        const v16h bO = *(const v16h*)(wtO + s * 512);
        const v16h bC = *(const v16h*)(wtC + s * 512);
        acO0 = WMMA_F16(a0[s], bO, acO0);
        acO1 = WMMA_F16(a1[s], bO, acO1);
        acC0 = WMMA_F16(a0[s], bC, acC0);
        acC1 = WMMA_F16(a1[s], bC, acC1);
      }

      // --- Epilogue: sigmoid via tanh identity, 3 TRANS/element.
      //   sigmoid(v+b) = 0.5 + 0.5*tanh(0.5*v + 0.5*b)
      const float hbo = ldsB[t * 16 + lh];
      const float hbc = ldsB[t * 16 + lh + 256];
      float* __restrict__ op0 =
          out + (size_t)(ch * 32 + hi8) * 256 + (size_t)(t * 16 + lh);
      float* __restrict__ op1 = op0 + (size_t)16 * 256;
#pragma unroll
      for (int v = 0; v < 8; ++v) {
        float og0 = __builtin_fmaf(0.5f, hw_tanh(__builtin_fmaf(0.5f, acO0[v], hbo)), 0.5f);
        float cg0 = __builtin_fmaf(0.5f, hw_tanh(__builtin_fmaf(0.5f, acC0[v], hbc)), 0.5f);
        op0[(size_t)v * 256] = og0 * hw_tanh(cg0);
        float og1 = __builtin_fmaf(0.5f, hw_tanh(__builtin_fmaf(0.5f, acO1[v], hbo)), 0.5f);
        float cg1 = __builtin_fmaf(0.5f, hw_tanh(__builtin_fmaf(0.5f, acC1[v], hbc)), 0.5f);
        op1[(size_t)v * 256] = og1 * hw_tanh(cg1);
      }
    }
  }
}

extern "C" void kernel_launch(void* const* d_in, const int* in_sizes, int n_in,
                              void* d_out, int out_size, void* d_ws, size_t ws_size,
                              hipStream_t stream) {
  // setup_inputs() order:
  // 0:x 1:wf 2:uf 3:biasf 4:wi 5:ui 6:biasi 7:wo 8:uo 9:biaso 10:wc 11:uc 12:biasc
  const float* x     = (const float*)d_in[0];
  const float* wo    = (const float*)d_in[7];
  const float* biaso = (const float*)d_in[9];
  const float* wc    = (const float*)d_in[10];
  const float* biasc = (const float*)d_in[12];
  _Float16* pk = (_Float16*)d_ws;                 // 128 KB packed weights

  pack_weights_f16<<<256, 256, 0, stream>>>(wo, wc, pk);
  mdlstm_fused_wmma<<<GRIDX, 256, 0, stream>>>(x, pk, biaso, biasc, (float*)d_out);
}
